// CKConv_30245159699083
// MI455X (gfx1250) — matrix operand order repaired and checked by reading, not compiled, not run on hardware
//
#include <hip/hip_runtime.h>
#include <stdint.h>

#define LSEQ   8192
#define NFFT   16384
#define NBINS  8193
#define NPANE  8194      // NBINS + 1 pad panel (TDM pair-loads)
#define XSTR   8208      // per-sequence spectrum row stride (bins)
#define NCH    4096      // Co*Ci
#define PI_F   3.14159265358979323846f

typedef __bf16 bf16_t;
typedef __attribute__((ext_vector_type(16))) __bf16 v16bf;
typedef __attribute__((ext_vector_type(8)))  float  v8f;
typedef unsigned int u32x4 __attribute__((ext_vector_type(4)));
typedef int          i32x8 __attribute__((ext_vector_type(8)));
typedef int          i32x4 __attribute__((ext_vector_type(4)));

union ABf { v16bf v; uint32_t u[8]; uint16_t h[16]; };
union CFr { v8f   v; float    f[8]; };

__device__ __forceinline__ uint16_t f2bfu(float x) {
    uint32_t u = __float_as_uint(x);
    uint32_t r = u + 0x7FFFu + ((u >> 16) & 1u);   // round-to-nearest-even
    return (uint16_t)(r >> 16);
}

// Tensor Data Mover: 1-D DWORD tile, global -> LDS (D# per ISA 08 §8.3/8.4)
// amdgpu-toolchain (clang-23) exposes the 6-arg builtin:
//   (u32x4 g0, i32x8 g1, i32x4 g2, i32x4 g3, i32x8 extra, i32 cpol)
__device__ __forceinline__ void tdm_load_1d(uint32_t lds_off, const void* gptr,
                                            uint32_t ndwords) {
    uint64_t ga = (uint64_t)(uintptr_t)gptr;
    u32x4 g0;
    g0[0] = 1u;                                     // count=1, user mode
    g0[1] = lds_off;                                // lds_addr
    g0[2] = (uint32_t)ga;                           // global_addr[31:0]
    g0[3] = (uint32_t)(ga >> 32) | (2u << 30);      // global_addr[56:32] | type=2
    i32x8 g1;
    g1[0] = (int)(2u << 16);                        // wg_mask=0, data_size=4B
    g1[1] = (int)((ndwords & 0xFFFFu) << 16);       // tensor_dim0 lo16
    g1[2] = (int)((ndwords >> 16) & 0xFFFFu) | (int)(1u << 16); // dim0 hi | tensor_dim1=1
    g1[3] = (int)(ndwords << 16);                   // tile_dim0 (16b)
    g1[4] = 0;                                      // tile_dim1/2 unused
    g1[5] = (int)ndwords;                           // tensor_dim0_stride lo32
    g1[6] = 0;
    g1[7] = 0;
    i32x4 z4 = {0, 0, 0, 0};
    i32x8 z8 = {0, 0, 0, 0, 0, 0, 0, 0};
    __builtin_amdgcn_tensor_load_to_lds(g0, g1, z4, z4, z8, 0);
}

// ---------------------------------------------------------------------------
// Kernel 1: SIREN MLP -> conv kernel, stored transposed kbufT[j=o*64+i][t]
// ---------------------------------------------------------------------------
__global__ __launch_bounds__(256) void mlp_kernel(
    const float* __restrict__ w1, const float* __restrict__ b1,
    const float* __restrict__ w2, const float* __restrict__ b2,
    const float* __restrict__ w3, const float* __restrict__ b3,
    float* __restrict__ kbufT)
{
    __shared__ float    sh1[16][32];
    __shared__ uint16_t sh2[16][32];

    const int tid = threadIdx.x;
    const int t0  = blockIdx.x * 16;
    const int nbl = blockIdx.y * 128;

    for (int idx = tid; idx < 512; idx += 256) {
        int r = idx >> 5, j = idx & 31;
        float rel = -1.0f + 2.0f * (float)(t0 + r) / 8191.0f;
        sh1[r][j] = __sinf(30.0f * (rel * w1[j] + b1[j]));
    }
    __syncthreads();
    for (int idx = tid; idx < 512; idx += 256) {
        int r = idx >> 5, m = idx & 31;
        float acc = b2[m];
        #pragma unroll
        for (int j = 0; j < 32; ++j) acc += sh1[r][j] * w2[j * 32 + m];
        sh2[r][m] = f2bfu(__sinf(30.0f * acc));
    }
    __syncthreads();

    const int wave = tid >> 5, lane = tid & 31;
    const int n0 = nbl + wave * 16;
    const int np = lane & 15;
    const int hi = lane >> 4;

    ABf a;
    {
        const uint32_t* src = (const uint32_t*)&sh2[np][0];
        const int b0 = hi * 8;
        #pragma unroll
        for (int q = 0; q < 4; ++q) { a.u[q] = src[b0 + q]; a.u[4 + q] = src[b0 + 4 + q]; }
    }
    ABf b;
    {
        const int kh = hi * 16;
        #pragma unroll
        for (int e = 0; e < 16; ++e)
            b.h[e] = f2bfu(w3[(size_t)(kh + e) * NCH + n0 + np]);
    }
    CFr c;
    #pragma unroll
    for (int v = 0; v < 8; ++v) c.f[v] = 0.0f;
    c.v = __builtin_amdgcn_wmma_f32_16x16x32_bf16(false, a.v, false, b.v,
                                                  (short)0, c.v, false, false);
    const float bias3 = b3[n0 + np];
    float* dst = kbufT + (size_t)(n0 + np) * LSEQ + t0 + hi * 8;
    #pragma unroll
    for (int v = 0; v < 8; ++v) dst[v] = c.f[v] + bias3;
}

// ---------------------------------------------------------------------------
// Radix-4 DIT core, 16384 = 4^7 points (input pre-loaded digit-reversed)
// ---------------------------------------------------------------------------
__device__ __forceinline__ void cmulf(float ar, float ai, float br, float bi,
                                      float& cr, float& ci) {
    cr = ar * br - ai * bi;
    ci = ar * bi + ai * br;
}

__device__ __forceinline__ int rev4_14(int j) {
    unsigned s = __brev((unsigned)j) >> 18;                 // 14-bit reversal
    return (int)(((s & 0x2AAAu) >> 1) | ((s & 0x1555u) << 1)); // swap bit pairs
}

__device__ void fft_stages(float* re, float* im)
{
    const int tid = threadIdx.x;
    for (int s = 0; s < 7; ++s) {
        const int   m    = 1 << (2 * s);
        const float wmul = -1.5707963267948966f / (float)m;   // -pi/2 / m
        for (int k = tid; k < NFFT / 4; k += 256) {
            int pos  = k & (m - 1);
            int base = ((k >> (2 * s)) << (2 * s + 2)) + pos;
            int i0 = base, i1 = base + m, i2 = base + 2 * m, i3 = base + 3 * m;
            float c1, s1;
            __sincosf(wmul * (float)pos, &s1, &c1);
            float c2 = c1 * c1 - s1 * s1, s2 = 2.0f * c1 * s1;
            float c3, s3; cmulf(c2, s2, c1, s1, c3, s3);
            float ar = re[i0], ai = im[i0];
            float br, bi; cmulf(re[i1], im[i1], c1, s1, br, bi);
            float cr, ci; cmulf(re[i2], im[i2], c2, s2, cr, ci);
            float dr, di; cmulf(re[i3], im[i3], c3, s3, dr, di);
            float t0r = ar + cr, t0i = ai + ci;
            float t1r = ar - cr, t1i = ai - ci;
            float t2r = br + dr, t2i = bi + di;
            float t3r = bi - di, t3i = -(br - dr);            // (b-d) * (-i)
            re[i0] = t0r + t2r; im[i0] = t0i + t2i;
            re[i1] = t1r + t3r; im[i1] = t1i + t3i;
            re[i2] = t0r - t2r; im[i2] = t0i - t2i;
            re[i3] = t1r - t3r; im[i3] = t1i - t3i;
        }
        __syncthreads();
    }
}

// ---------------------------------------------------------------------------
// Kernel 2: forward rFFT of x (zero-padded), bf16 spectrum out
// ---------------------------------------------------------------------------
__global__ __launch_bounds__(256) void fft_x_kernel(
    const float* __restrict__ x, uint16_t* __restrict__ Xs)
{
    extern __shared__ float sm[];
    float* re = sm; float* im = sm + NFFT;
    const int s = blockIdx.x;                       // b*64+i
    const float* row = x + (size_t)s * LSEQ;
    for (int j = threadIdx.x; j < NFFT; j += 256) {
        int r = rev4_14(j);
        re[j] = (r < LSEQ) ? row[r] : 0.0f;
        im[j] = 0.0f;
    }
    __syncthreads();
    fft_stages(re, im);
    uint16_t* out = Xs + (size_t)s * XSTR * 2;
    for (int f = threadIdx.x; f < NBINS; f += 256) {
        out[2 * f]     = f2bfu(re[f]);
        out[2 * f + 1] = f2bfu(im[f]);
    }
}

// ---------------------------------------------------------------------------
// Kernel 3: forward rFFT of time-reversed kernel rows (kr), bf16 spectrum out
// ---------------------------------------------------------------------------
__global__ __launch_bounds__(256) void fft_k_kernel(
    const float* __restrict__ kbufT, uint16_t* __restrict__ Ks)
{
    extern __shared__ float sm[];
    float* re = sm; float* im = sm + NFFT;
    const int s = blockIdx.x;                       // o*64+i
    const float* row = kbufT + (size_t)s * LSEQ;
    for (int j = threadIdx.x; j < NFFT; j += 256) {
        int r = rev4_14(j);
        re[j] = (r < LSEQ) ? row[(LSEQ - 1) - r] : 0.0f;   // kr = k reversed
        im[j] = 0.0f;
    }
    __syncthreads();
    fft_stages(re, im);
    uint16_t* out = Ks + (size_t)s * XSTR * 2;
    for (int f = threadIdx.x; f < NBINS; f += 256) {
        out[2 * f]     = f2bfu(re[f]);
        out[2 * f + 1] = f2bfu(im[f]);
    }
}

// ---------------------------------------------------------------------------
// Kernel 4: pack X spectra -> WMMA A layout  Xp[f][m(16)][128 = Re_i|Im_i]
// ---------------------------------------------------------------------------
__global__ __launch_bounds__(256) void pack_x_kernel(
    const uint16_t* __restrict__ Xs, uint16_t* __restrict__ Xp)
{
    const int f = blockIdx.x;
    for (int idx = threadIdx.x; idx < 2048; idx += 256) {
        int m = idx >> 7, p = idx & 127;
        uint16_t v = 0;
        if (m < 8) {
            int i = p & 63, tau = p >> 6;
            v = Xs[((size_t)(m * 64 + i) * XSTR + f) * 2 + tau];
        }
        Xp[(size_t)f * 2048 + idx] = v;
    }
}

// ---------------------------------------------------------------------------
// Kernel 5: pack K spectra -> Kp[f][tau][o*64+i]  (tiled (s,f) transpose)
// ---------------------------------------------------------------------------
__global__ __launch_bounds__(256) void pack_k_kernel(
    const uint16_t* __restrict__ Ks, uint16_t* __restrict__ Kp)
{
    __shared__ uint16_t tile[256][2][34];
    const int f0 = blockIdx.x * 32;
    const int s0 = blockIdx.y * 256;
    const int ts = threadIdx.x;
    const uint16_t* src = Ks + ((size_t)(s0 + ts) * XSTR + f0) * 2;
    for (int q = 0; q < 32; ++q) {
        uint16_t vr = 0, vi = 0;
        if (f0 + q < NBINS) { vr = src[2 * q]; vi = src[2 * q + 1]; }
        tile[ts][0][q] = vr; tile[ts][1][q] = vi;
    }
    __syncthreads();
    for (int idx = threadIdx.x; idx < 16384; idx += 256) {
        int fl  = idx >> 9;
        int tau = (idx >> 8) & 1;
        int sl  = idx & 255;
        int f   = f0 + fl;
        if (f < NBINS)
            Kp[(size_t)f * 8192 + tau * 4096 + s0 + sl] = tile[sl][tau][fl];
    }
}

// ---------------------------------------------------------------------------
// Kernel 6: per-frequency complex contraction Y[b,o] = sum_i X[b,i]*K[o,i]
//   TDM stages 2 frequencies' X/K panels to LDS; each wave computes Yr AND Yi
//   for one (freq, o-tile): 4 A frags + 4 B frags reused across 8 WMMAs
// ---------------------------------------------------------------------------
__global__ __launch_bounds__(256) void einsum_kernel(
    const uint16_t* __restrict__ Xp, const uint16_t* __restrict__ Kp,
    float* __restrict__ Yre, float* __restrict__ Yim)
{
    __shared__ uint16_t sKp[2 * 8192];   // 32 KB: K panels f0, f0+1
    __shared__ uint16_t sXp[2 * 2048];   //  8 KB: X panels f0, f0+1

    const int f0  = blockIdx.x * 2;
    const int tid = threadIdx.x;

    if (tid < 32) {   // wave 0 issues the DMAs
        tdm_load_1d((uint32_t)(uintptr_t)&sKp[0], Kp + (size_t)f0 * 8192, 8192);
        tdm_load_1d((uint32_t)(uintptr_t)&sXp[0], Xp + (size_t)f0 * 2048, 2048);
        __builtin_amdgcn_s_wait_tensorcnt(0);
    }
    __syncthreads();
    asm volatile("" ::: "memory");

    const int wave = tid >> 5, lane = tid & 31;
    const int fi = wave >> 2;            // frequency within pair
    const int o0 = (wave & 3) * 16;      // o-tile
    const int np = lane & 15;
    const int hi = lane >> 4;
    const int f  = f0 + fi;

    const uint16_t* Arow = sXp + fi * 2048 + np * 128;
    const uint16_t* Kf   = sKp + fi * 8192;

    ABf A[4];
    #pragma unroll
    for (int kc = 0; kc < 4; ++kc) {
        const int abase = kc * 32 + hi * 16;
        const uint32_t* a0 = (const uint32_t*)(Arow + abase);
        const uint32_t* a1 = (const uint32_t*)(Arow + abase + 8);
        #pragma unroll
        for (int q = 0; q < 4; ++q) { A[kc].u[q] = a0[q]; A[kc].u[4 + q] = a1[q]; }
    }
    ABf B[2][2], Bn[2];                  // B[tau][ihalf]; Bn = -B[1][*]
    #pragma unroll
    for (int tau = 0; tau < 2; ++tau)
        #pragma unroll
        for (int h = 0; h < 2; ++h) {
            const uint32_t* bs =
                (const uint32_t*)(Kf + tau * 4096 + (o0 + np) * 64 + h * 32 + hi * 16);
            #pragma unroll
            for (int q = 0; q < 8; ++q) B[tau][h].u[q] = bs[q];
        }
    #pragma unroll
    for (int h = 0; h < 2; ++h)
        #pragma unroll
        for (int q = 0; q < 8; ++q) Bn[h].u[q] = B[1][h].u[q] ^ 0x80008000u;

    CFr accR, accI;
    #pragma unroll
    for (int v = 0; v < 8; ++v) { accR.f[v] = 0.0f; accI.f[v] = 0.0f; }

    // Yr = Xr*KrT - Xi*KiT   (A0,A1 = Xr chunks; A2,A3 = Xi chunks)
    accR.v = __builtin_amdgcn_wmma_f32_16x16x32_bf16(false, A[0].v, false, B[0][0].v, (short)0, accR.v, false, false);
    accR.v = __builtin_amdgcn_wmma_f32_16x16x32_bf16(false, A[1].v, false, B[0][1].v, (short)0, accR.v, false, false);
    accR.v = __builtin_amdgcn_wmma_f32_16x16x32_bf16(false, A[2].v, false, Bn[0].v,   (short)0, accR.v, false, false);
    accR.v = __builtin_amdgcn_wmma_f32_16x16x32_bf16(false, A[3].v, false, Bn[1].v,   (short)0, accR.v, false, false);
    // Yi = Xr*KiT + Xi*KrT
    accI.v = __builtin_amdgcn_wmma_f32_16x16x32_bf16(false, A[0].v, false, B[1][0].v, (short)0, accI.v, false, false);
    accI.v = __builtin_amdgcn_wmma_f32_16x16x32_bf16(false, A[1].v, false, B[1][1].v, (short)0, accI.v, false, false);
    accI.v = __builtin_amdgcn_wmma_f32_16x16x32_bf16(false, A[2].v, false, B[0][0].v, (short)0, accI.v, false, false);
    accI.v = __builtin_amdgcn_wmma_f32_16x16x32_bf16(false, A[3].v, false, B[0][1].v, (short)0, accI.v, false, false);

    if (lane < 16 && f < NBINS) {        // rows M=0..7 are the valid batch rows
        float* dstR = Yre + (size_t)f * 512;
        float* dstI = Yim + (size_t)f * 512;
        #pragma unroll
        for (int v = 0; v < 8; ++v) {
            dstR[v * 64 + o0 + lane] = accR.f[v];
            dstI[v * 64 + o0 + lane] = accI.f[v];
        }
    }
}

// ---------------------------------------------------------------------------
// Kernel 7: irFFT (Hermitian rebuild + conj-trick inverse) + bias
// ---------------------------------------------------------------------------
__global__ __launch_bounds__(256) void ifft_kernel(
    const float* __restrict__ Yre, const float* __restrict__ Yim,
    const float* __restrict__ bias, float* __restrict__ y)
{
    extern __shared__ float sm[];
    float* re = sm; float* im = sm + NFFT;
    const int s = blockIdx.x;            // b*64+o
    const int o = s & 63;
    for (int j = threadIdx.x; j < NFFT; j += 256) {
        int r = rev4_14(j);
        int src; float sgn;
        if (r <= 8192) { src = r;        sgn = -1.0f; }  // conj(Y[r])
        else           { src = NFFT - r; sgn =  1.0f; }  // conj(conj(Y)) = Y
        re[j] = Yre[(size_t)src * 512 + s];
        im[j] = sgn * Yim[(size_t)src * 512 + s];
    }
    __syncthreads();
    fft_stages(re, im);
    const float bo = bias[o];
    float* out = y + (size_t)s * LSEQ;
    for (int t = threadIdx.x; t < LSEQ; t += 256)
        out[t] = re[t] * (1.0f / 16384.0f) + bo;
}

// ---------------------------------------------------------------------------
extern "C" void kernel_launch(void* const* d_in, const int* in_sizes, int n_in,
                              void* d_out, int out_size, void* d_ws, size_t ws_size,
                              hipStream_t stream)
{
    const float* x    = (const float*)d_in[0];
    const float* w1   = (const float*)d_in[1];
    const float* b1   = (const float*)d_in[2];
    const float* w2   = (const float*)d_in[3];
    const float* b2   = (const float*)d_in[4];
    const float* w3   = (const float*)d_in[5];
    const float* b3   = (const float*)d_in[6];
    const float* bias = (const float*)d_in[7];

    uint8_t* ws = (uint8_t*)d_ws;
    float*    kbufT = (float*)   (ws);                // 134,217,728 B
    uint16_t* Xs    = (uint16_t*)(ws + 134217728);    //  16,809,984 B
    uint16_t* Ks    = (uint16_t*)(ws + 151027712);    // 134,479,872 B
    uint16_t* Xp    = (uint16_t*)(ws + 285507584);    //  33,562,624 B (NPANE panels)
    uint16_t* Kp    = (uint16_t*)(ws + 319070208);    // 134,250,496 B (NPANE panels)
    float*    Yre   = (float*)   (ws + 453320704);    //  16,779,264 B
    float*    Yim   = (float*)   (ws + 470099968);    //  16,779,264 B

    // 128 KB dynamic LDS for the FFT kernels (WGP allows up to 320 KB)
    (void)hipFuncSetAttribute((const void*)fft_x_kernel,
            hipFuncAttributeMaxDynamicSharedMemorySize, 131072);
    (void)hipFuncSetAttribute((const void*)fft_k_kernel,
            hipFuncAttributeMaxDynamicSharedMemorySize, 131072);
    (void)hipFuncSetAttribute((const void*)ifft_kernel,
            hipFuncAttributeMaxDynamicSharedMemorySize, 131072);

    mlp_kernel<<<dim3(LSEQ / 16, NCH / 128), 256, 0, stream>>>(w1, b1, w2, b2, w3, b3, kbufT);
    fft_x_kernel<<<dim3(512),  256, 131072, stream>>>(x, Xs);
    fft_k_kernel<<<dim3(NCH),  256, 131072, stream>>>(kbufT, Ks);
    pack_x_kernel<<<dim3(NBINS), 256, 0, stream>>>(Xs, Xp);
    pack_k_kernel<<<dim3((NBINS + 31) / 32, 16), 256, 0, stream>>>(Ks, Kp);
    einsum_kernel<<<dim3((NBINS + 1) / 2), 256, 0, stream>>>(Xp, Kp, Yre, Yim);
    ifft_kernel<<<dim3(512), 256, 131072, stream>>>(Yre, Yim, bias, (float*)d_out);
}
